// SAGEConvModel_17635135718039
// MI455X (gfx1250) — matrix operand order repaired and checked by reading, not compiled, not run on hardware
//
#include <hip/hip_runtime.h>
#include <hip/hip_bf16.h>
#include <math.h>

typedef __attribute__((ext_vector_type(16))) _Float16 v16h;
typedef __attribute__((ext_vector_type(8)))  float    v8f;

#define NNODES 100000
#define NEDGES 1600000
#define FEAT   32        // padded channel stride for all node buffers
#define NGRAPH 512
#define CH     30
#define BN_EPS 1e-5f

// ---------------------------------------------------------------------------
// Workspace init: zero accumulators, -inf the max-pool buffer.
// ---------------------------------------------------------------------------
__global__ void init_ws(float* __restrict__ cnt, float* __restrict__ s1,
                        float* __restrict__ s2, unsigned* __restrict__ pmax,
                        float* __restrict__ psum, float* __restrict__ pcnt) {
  long long i = (long long)blockIdx.x * blockDim.x + threadIdx.x;
  long long stride = (long long)gridDim.x * blockDim.x;
  for (long long j = i; j < (long long)NNODES; j += stride) cnt[j] = 0.f;
  for (long long j = i; j < (long long)NNODES * FEAT; j += stride) {
    s1[j] = 0.f;
    s2[j] = 0.f;
  }
  for (long long j = i; j < (long long)NGRAPH * CH; j += stride) {
    pmax[j] = 0xFF800000u;  // -inf
    psum[j] = 0.f;
  }
  for (long long j = i; j < (long long)NGRAPH; j += stride) pcnt[j] = 0.f;
}

// ---------------------------------------------------------------------------
// Pre-pack weights into the exact WMMA B-matrix lane layout (f16, zero-padded)
// and fold BatchNorm into per-channel scale/shift tables (zero-padded to 32).
//
// Packed B layout: pb[((layer*2 + t)*2 + mat)*512 + lane*16 + i]
//   B (32x16 f16): n = (lane&15) + 16*t, k = (i&7) + 8*(lane>>4) + 16*(i>>3)
//   mat: 0 = Wl (applied to agg), 1 = Wr (applied to x)
// ---------------------------------------------------------------------------
__global__ void pack_params(const float* __restrict__ Wl1, const float* __restrict__ bl1,
                            const float* __restrict__ Wr1,
                            const float* __restrict__ Wl2, const float* __restrict__ bl2,
                            const float* __restrict__ Wr2,
                            const float* __restrict__ g1, const float* __restrict__ b1,
                            const float* __restrict__ m1, const float* __restrict__ v1,
                            const float* __restrict__ g2, const float* __restrict__ b2,
                            const float* __restrict__ m2, const float* __restrict__ v2,
                            _Float16* __restrict__ pb, float* __restrict__ pbias,
                            float* __restrict__ pscale, float* __restrict__ pshift) {
  int tid = blockIdx.x * blockDim.x + threadIdx.x;
  if (tid < 4096) {
    int i     = tid & 15;
    int lane  = (tid >> 4) & 31;
    int mat   = (tid >> 9) & 1;
    int t     = (tid >> 10) & 1;
    int layer = (tid >> 11) & 1;
    int n = (lane & 15) + 16 * t;
    int k = (i & 7) + 8 * (lane >> 4) + 16 * (i >> 3);
    int kdim = layer ? CH : FEAT;
    const float* W = layer ? (mat ? Wr2 : Wl2) : (mat ? Wr1 : Wl1);
    float w = (n < CH && k < kdim) ? W[n * kdim + k] : 0.f;
    pb[tid] = (_Float16)w;
  }
  if (tid < 64) {
    int layer = tid >> 5;
    int n = tid & 31;
    const float* blp = layer ? bl2 : bl1;
    const float* gp  = layer ? g2 : g1;
    const float* bp  = layer ? b2 : b1;
    const float* mp  = layer ? m2 : m1;
    const float* vp  = layer ? v2 : v1;
    float bias = 0.f, scale = 0.f, shift = 0.f;
    if (n < CH) {
      float rs = rsqrtf(vp[n] + BN_EPS);
      scale = gp[n] * rs;
      shift = bp[n] - mp[n] * scale;
      bias  = blp[n];
    }
    pbias[tid]  = bias;
    pscale[tid] = scale;
    pshift[tid] = shift;
  }
}

// ---------------------------------------------------------------------------
// Edge scatter-add: s[dst] += feat[src]; optional degree count.
// 4 threads/edge, each: 2x float4 gather + 8 float atomics (L2-resident).
// ---------------------------------------------------------------------------
__global__ void edge_agg(const int* __restrict__ src, const int* __restrict__ dst,
                         const float* __restrict__ feat, float* __restrict__ s,
                         float* __restrict__ cnt) {
  long long idx = (long long)blockIdx.x * blockDim.x + threadIdx.x;
  long long total = (long long)NEDGES * 4;
  long long stride = (long long)gridDim.x * blockDim.x;
  for (long long t = idx; t < total; t += stride) {
    int e = (int)(t >> 2);
    int q = (int)(t & 3);           // 8-feature octet
    int se = src[e];
    int de = dst[e];
    const float4* pv = (const float4*)(feat + (long long)se * FEAT + q * 8);
    float4 v0 = pv[0];
    float4 v1 = pv[1];
    float* p = s + (long long)de * FEAT + q * 8;
    atomicAdd(p + 0, v0.x);
    atomicAdd(p + 1, v0.y);
    atomicAdd(p + 2, v0.z);
    atomicAdd(p + 3, v0.w);
    atomicAdd(p + 4, v1.x);
    atomicAdd(p + 5, v1.y);
    atomicAdd(p + 6, v1.z);
    atomicAdd(p + 7, v1.w);
    if (cnt != nullptr && q == 0) atomicAdd(cnt + de, 1.0f);
  }
}

// ---------------------------------------------------------------------------
// Per-node SAGE transform, branch-free WMMA:
//   h = BN(ReLU( (s/cnt) @ Wl^T + bl + x @ Wr^T ))
// One wave = one 16-node tile; 4x v_wmma_f32_16x16x32_f16 per tile.
// A loads: 4x b128 per matrix. B loads: packed v16h (2x b128) per operand.
// ---------------------------------------------------------------------------
__global__ void __launch_bounds__(256) sage_node(
    const float* __restrict__ s, const float* __restrict__ cnt,
    const float* __restrict__ xin,
    const _Float16* __restrict__ pb,    // this layer's 2048 packed halves
    const float* __restrict__ pbias,    // [32]
    const float* __restrict__ pscale,   // [32]
    const float* __restrict__ pshift,   // [32]
    float* __restrict__ hout) {
  const int lane = threadIdx.x & 31;
  const int wave = threadIdx.x >> 5;
  const int tile = blockIdx.x * 8 + wave;
  const int base = tile * 16;
  if (base >= NNODES) return;  // uniform per wave -> EXEC all-1s for WMMA

  const int g  = lane >> 4;   // lane group
  const int mn = lane & 15;   // M for A-matrix, N for B-matrix
  const int row = base + mn;

  const float inv = 1.0f / fmaxf(cnt[row], 1.0f);

  // A matrices (16x32 f16). K footprint per lane: [8g,8g+8) and [16+8g,24+8g).
  const float* ps = s   + (long long)row * FEAT + (g << 3);
  const float* px = xin + (long long)row * FEAT + (g << 3);
  float4 sa0 = *(const float4*)(ps);
  float4 sa1 = *(const float4*)(ps + 4);
  float4 sa2 = *(const float4*)(ps + 16);
  float4 sa3 = *(const float4*)(ps + 20);
  float4 xa0 = *(const float4*)(px);
  float4 xa1 = *(const float4*)(px + 4);
  float4 xa2 = *(const float4*)(px + 16);
  float4 xa3 = *(const float4*)(px + 20);

  float sbuf[16] = {sa0.x, sa0.y, sa0.z, sa0.w, sa1.x, sa1.y, sa1.z, sa1.w,
                    sa2.x, sa2.y, sa2.z, sa2.w, sa3.x, sa3.y, sa3.z, sa3.w};
  float xbuf[16] = {xa0.x, xa0.y, xa0.z, xa0.w, xa1.x, xa1.y, xa1.z, xa1.w,
                    xa2.x, xa2.y, xa2.z, xa2.w, xa3.x, xa3.y, xa3.z, xa3.w};
  v16h Aagg, Ax;
#pragma unroll
  for (int i = 0; i < 16; ++i) {
    Aagg[i] = (_Float16)(sbuf[i] * inv);
    Ax[i]   = (_Float16)(xbuf[i]);
  }

  v8f acc[2];
#pragma unroll
  for (int t = 0; t < 2; ++t) {
    const _Float16* p = pb + t * 1024 + lane * 16;
    v16h Bl = *(const v16h*)(p);        // mat 0 (Wl)
    v16h Br = *(const v16h*)(p + 512);  // mat 1 (Wr)
    const float bias = pbias[mn + 16 * t];
    v8f c;
#pragma unroll
    for (int r = 0; r < 8; ++r) c[r] = bias;
    c = __builtin_amdgcn_wmma_f32_16x16x32_f16(false, Aagg, false, Bl,
                                               (short)0, c, false, false);
    c = __builtin_amdgcn_wmma_f32_16x16x32_f16(false, Ax, false, Br,
                                               (short)0, c, false, false);
    acc[t] = c;
  }

  // Epilogue: ReLU + folded BatchNorm; padded channels give exact 0.
#pragma unroll
  for (int t = 0; t < 2; ++t) {
    const int n = mn + 16 * t;
    const float scale = pscale[n];
    const float shift = pshift[n];
#pragma unroll
    for (int r = 0; r < 8; ++r) {
      int orow = base + r + 8 * g;  // D layout: VGPR r -> M = r + 8*laneGroup
      float val = fmaxf(acc[t][r], 0.f) * scale + shift;
      hout[(long long)orow * FEAT + n] = val;
    }
  }
}

// ---------------------------------------------------------------------------
// Dual segment pooling (sum + max) into per-graph buffers via L2 atomics.
// ---------------------------------------------------------------------------
__global__ void pool_kernel(const float* __restrict__ h, const int* __restrict__ batch,
                            unsigned* __restrict__ pmax, float* __restrict__ psum,
                            float* __restrict__ pcnt) {
  long long idx = (long long)blockIdx.x * blockDim.x + threadIdx.x;
  long long total = (long long)NNODES * FEAT;
  long long stride = (long long)gridDim.x * blockDim.x;
  for (long long t = idx; t < total; t += stride) {
    int node = (int)(t >> 5);
    int c    = (int)(t & 31);
    int gph  = batch[node];
    if (c == 0) atomicAdd(pcnt + gph, 1.0f);
    if (c < CH) {
      float val = h[t];
      atomicAdd(psum + (long long)gph * CH + c, val);
      unsigned* mp = pmax + (long long)gph * CH + c;
      if (val >= 0.f)
        atomicMax((int*)mp, __float_as_int(val));
      else
        atomicMin(mp, __float_as_uint(val));
    }
  }
}

// ---------------------------------------------------------------------------
// Final MLP head: [max || mean] -> 10 (ReLU) -> 1 -> sigmoid. One thread/graph.
// ---------------------------------------------------------------------------
__global__ void mlp_kernel(const unsigned* __restrict__ pmax, const float* __restrict__ psum,
                           const float* __restrict__ pcnt,
                           const float* __restrict__ W1, const float* __restrict__ bb1,
                           const float* __restrict__ W2, const float* __restrict__ bb2,
                           float* __restrict__ out) {
  int g = blockIdx.x * blockDim.x + threadIdx.x;
  if (g >= NGRAPH) return;
  float z[2 * CH];
  float invc = 1.0f / fmaxf(pcnt[g], 1.0f);
  for (int c = 0; c < CH; ++c) {
    float mv = __uint_as_float(pmax[g * CH + c]);
    z[c]      = isfinite(mv) ? mv : 0.f;  // empty graphs: -inf -> 0
    z[CH + c] = psum[g * CH + c] * invc;
  }
  float o = bb2[0];
  for (int i = 0; i < 10; ++i) {
    float a = bb1[i];
    for (int j = 0; j < 2 * CH; ++j) a += W1[i * 2 * CH + j] * z[j];
    o += W2[i] * fmaxf(a, 0.f);
  }
  out[g] = 1.0f / (1.0f + expf(-o));
}

// ---------------------------------------------------------------------------
extern "C" void kernel_launch(void* const* d_in, const int* in_sizes, int n_in,
                              void* d_out, int out_size, void* d_ws, size_t ws_size,
                              hipStream_t stream) {
  const float* x   = (const float*)d_in[0];
  const int*   ei  = (const int*)d_in[1];
  const int*   bat = (const int*)d_in[2];
  const float* Wl1 = (const float*)d_in[3];
  const float* bl1 = (const float*)d_in[4];
  const float* Wr1 = (const float*)d_in[5];
  const float* Wl2 = (const float*)d_in[6];
  const float* bl2 = (const float*)d_in[7];
  const float* Wr2 = (const float*)d_in[8];
  const float* g1  = (const float*)d_in[9];
  const float* b1  = (const float*)d_in[10];
  const float* m1  = (const float*)d_in[11];
  const float* v1  = (const float*)d_in[12];
  const float* g2  = (const float*)d_in[13];
  const float* b2  = (const float*)d_in[14];
  const float* m2  = (const float*)d_in[15];
  const float* v2  = (const float*)d_in[16];
  const float* W1  = (const float*)d_in[17];
  const float* bb1 = (const float*)d_in[18];
  const float* W2  = (const float*)d_in[19];
  const float* bb2 = (const float*)d_in[20];

  const int* src = ei;            // edge_index[0]
  const int* dst = ei + NEDGES;   // edge_index[1]

  // Workspace layout (all offsets 32B-aligned):
  //   cnt | bufA | bufB | bufC | pmax | psum | pcnt | packB | pbias|pscale|pshift
  //   bufA: layer-1 agg sums, later reused as h2 output
  //   bufB: h1 ; bufC: layer-2 agg sums
  float*    ws    = (float*)d_ws;
  float*    cnt   = ws;
  float*    bufA  = cnt  + NNODES;
  float*    bufB  = bufA + (size_t)NNODES * FEAT;
  float*    bufC  = bufB + (size_t)NNODES * FEAT;
  unsigned* pmax  = (unsigned*)(bufC + (size_t)NNODES * FEAT);
  float*    psum  = (float*)(pmax + (size_t)NGRAPH * CH);
  float*    pcnt  = psum + (size_t)NGRAPH * CH;
  _Float16* packB = (_Float16*)(pcnt + NGRAPH);      // 4096 halves (8 KB)
  float*    pbias = (float*)(packB + 4096);          // [2][32]
  float*    pscale = pbias + 64;                     // [2][32]
  float*    pshift = pscale + 64;                    // [2][32]

  const int sage_blocks = ((NNODES / 16) + 7) / 8;  // 8 waves (tiles) per block

  init_ws<<<2048, 256, 0, stream>>>(cnt, bufA, bufC, pmax, psum, pcnt);
  pack_params<<<16, 256, 0, stream>>>(Wl1, bl1, Wr1, Wl2, bl2, Wr2,
                                      g1, b1, m1, v1, g2, b2, m2, v2,
                                      packB, pbias, pscale, pshift);
  // Layer 1
  edge_agg<<<4096, 256, 0, stream>>>(src, dst, x, bufA, cnt);
  sage_node<<<sage_blocks, 256, 0, stream>>>(bufA, cnt, x,
                                             packB, pbias, pscale, pshift, bufB);
  // Layer 2 (degree reused)
  edge_agg<<<4096, 256, 0, stream>>>(src, dst, bufB, bufC, nullptr);
  sage_node<<<sage_blocks, 256, 0, stream>>>(bufC, cnt, bufB,
                                             packB + 2048, pbias + 32,
                                             pscale + 32, pshift + 32, bufA);
  // Pool + head
  pool_kernel<<<4096, 256, 0, stream>>>(bufA, bat, pmax, psum, pcnt);
  mlp_kernel<<<(NGRAPH + 255) / 256, 256, 0, stream>>>(pmax, psum, pcnt,
                                                       W1, bb1, W2, bb2,
                                                       (float*)d_out);
}
